// YOLOgDSATransformerVisualOnly_19885698580774
// MI455X (gfx1250) — compile-verified, matrix operand order checked by reading, not run on hardware
//
#include <hip/hip_runtime.h>
#include <hip/hip_bf16.h>

// ---------------------------------------------------------------------------
// CDNA5 (gfx1250) graph-transformer implementation.
// All GEMMs run through v_wmma_f32_16x16x32_bf16 (wave32, 2x2 16x16 tiles/wave,
// b128 fragment loads, clamped B rows -> no EXEC divergence in the K loop).
// ---------------------------------------------------------------------------

typedef __attribute__((ext_vector_type(16))) __bf16 v16bf;
typedef __attribute__((ext_vector_type(8)))  float  v8f;

#define GN      2048
#define GE      65536
#define GHID    256
#define GEDGE   64
#define GHEADS  8
#define GHD     32
#define GLAYERS 4

__device__ __forceinline__ unsigned short f2bf(float f) {
  unsigned int x = __float_as_uint(f);
  x += 0x7fffu + ((x >> 16) & 1u);        // round-to-nearest-even
  return (unsigned short)(x >> 16);
}
__device__ __forceinline__ float gelu_f(float x) {
  return 0.5f * x * (1.0f + erff(x * 0.70710678118654752f));   // exact GELU
}

union BFrag { v16bf v; uint4 q[2]; };

enum { FBIAS = 1, FGELU = 2, FRES = 4, FF32 = 8, FBF16 = 16 };

// ---------------------------------------------------------------------------
// Generic bf16 WMMA GEMM: C[M,Nc] = alpha * A[M,K] * Bt[Nc,K]^T (+bias,res,gelu)
// A row-major bf16, Bt = B^T row-major bf16. Fragment layout (16-bit A-style):
// lane half h, row r=lane&15; VGPR 0..3 -> K = h*8 + 0..7 (contiguous, 16B),
// VGPR 4..7 -> K = 16 + h*8 + 0..7 (contiguous, 16B) => two b128 loads/frag.
// Each wave: 32x32 output (2x2 WMMA tiles). 8 waves/block stacked in M.
// blockIdx.z batches heads via sAz/sBz/sCz strides.
// ---------------------------------------------------------------------------
__global__ __launch_bounds__(256) void k_gemm(
    const unsigned short* __restrict__ A, int lda, unsigned long long sAz,
    const unsigned short* __restrict__ Bt, int ldb, unsigned long long sBz,
    float* __restrict__ Cf, unsigned short* __restrict__ Cb, int ldc,
    unsigned long long sCz,
    const float* __restrict__ bias, const float* __restrict__ resid,
    float alpha, int M, int Ncols, int K, int flags)
{
  A  += sAz * blockIdx.z;
  Bt += sBz * blockIdx.z;
  const int lane = threadIdx.x & 31;
  const int wave = threadIdx.x >> 5;
  const int m0 = (blockIdx.y * 8 + wave) * 32;
  const int n0 = blockIdx.x * 32;
  if (m0 >= M) return;                    // M is a multiple of 32 in all uses
  const int half = lane >> 4;
  const int r    = lane & 15;

  v8f c00 = {}, c01 = {}, c10 = {}, c11 = {};

  const unsigned short* a0p = A + (size_t)(m0 + r)      * lda + half * 8;
  const unsigned short* a1p = A + (size_t)(m0 + 16 + r) * lda + half * 8;
  // clamp B rows: an OOB row only feeds output col bn, which the store masks
  const int bn0 = min(n0 + r,      Ncols - 1);
  const int bn1 = min(n0 + 16 + r, Ncols - 1);
  const unsigned short* b0p = Bt + (size_t)bn0 * ldb + half * 8;
  const unsigned short* b1p = Bt + (size_t)bn1 * ldb + half * 8;

  for (int k = 0; k < K; k += 32) {
    BFrag a0, a1, b0, b1;
    a0.q[0] = *(const uint4*)(a0p + k);
    a0.q[1] = *(const uint4*)(a0p + k + 16);
    a1.q[0] = *(const uint4*)(a1p + k);
    a1.q[1] = *(const uint4*)(a1p + k + 16);
    b0.q[0] = *(const uint4*)(b0p + k);
    b0.q[1] = *(const uint4*)(b0p + k + 16);
    b1.q[0] = *(const uint4*)(b1p + k);
    b1.q[1] = *(const uint4*)(b1p + k + 16);
    if (k + 32 < K) {                     // lowers to global_prefetch_b8
      __builtin_prefetch(a0p + k + 32, 0, 0);
      __builtin_prefetch(b0p + k + 32, 0, 0);
    }
    c00 = __builtin_amdgcn_wmma_f32_16x16x32_bf16(false, a0.v, false, b0.v, (short)0, c00, false, false);
    c01 = __builtin_amdgcn_wmma_f32_16x16x32_bf16(false, a0.v, false, b1.v, (short)0, c01, false, false);
    c10 = __builtin_amdgcn_wmma_f32_16x16x32_bf16(false, a1.v, false, b0.v, (short)0, c10, false, false);
    c11 = __builtin_amdgcn_wmma_f32_16x16x32_bf16(false, a1.v, false, b1.v, (short)0, c11, false, false);
  }

  const size_t zoff = sCz * blockIdx.z;
  const int col0 = n0 + r, col1 = n0 + 16 + r;
  auto store_tile = [&](const v8f& c, int mbase, int col) {
    if (col >= Ncols) return;
    const float bv = (flags & FBIAS) ? bias[col] : 0.0f;
#pragma unroll
    for (int i = 0; i < 8; ++i) {
      const int row = mbase + half * 8 + i;   // C layout: VGPR i -> M=i (+8 hi half)
      float v = c[i] * alpha + bv;
      if (flags & FRES)  v += resid[(size_t)row * ldc + col];
      if (flags & FGELU) v = gelu_f(v);
      const size_t o = zoff + (size_t)row * ldc + col;
      if (flags & FF32)  Cf[o] = v;
      if (flags & FBF16) Cb[o] = f2bf(v);
    }
  };
  store_tile(c00, m0,      col0);
  store_tile(c01, m0,      col1);
  store_tile(c10, m0 + 16, col0);
  store_tile(c11, m0 + 16, col1);
}

// W[K,Nc] fp32 -> Bt[Nc,K] bf16 (transpose + convert)
__global__ void k_wconv(const float* __restrict__ W,
                        unsigned short* __restrict__ Bt, int K, int Nc) {
  int i = blockIdx.x * 256 + threadIdx.x;
  if (i >= K * Nc) return;
  int k = i / Nc, n = i % Nc;
  Bt[(size_t)n * K + k] = f2bf(W[i]);
}

__global__ void k_f32bf16(const float* __restrict__ x,
                          unsigned short* __restrict__ y, size_t n) {
  size_t i = (size_t)blockIdx.x * 256 + threadIdx.x;
  if (i < n) y[i] = f2bf(x[i]);
}

// LayerNorm over D (<=256), optional GELU, bf16 and/or fp32 outputs
__global__ __launch_bounds__(256) void k_ln(
    const float* __restrict__ X, const float* __restrict__ g,
    const float* __restrict__ b, unsigned short* __restrict__ Yb,
    float* __restrict__ Yf, int D, int dogelu)
{
  __shared__ float red[256];
  const int row = blockIdx.x, t = threadIdx.x;
  const float* x = X + (size_t)row * D;
  float v = (t < D) ? x[t] : 0.0f;
  red[t] = v; __syncthreads();
  for (int s = 128; s > 0; s >>= 1) { if (t < s) red[t] += red[t + s]; __syncthreads(); }
  const float mu = red[0] / D; __syncthreads();
  float d = (t < D) ? (v - mu) : 0.0f;
  red[t] = d * d; __syncthreads();
  for (int s = 128; s > 0; s >>= 1) { if (t < s) red[t] += red[t + s]; __syncthreads(); }
  const float var = red[0] / D;
  if (t < D) {
    float y = d * rsqrtf(var + 1e-5f) * g[t] + b[t];
    if (dogelu) y = gelu_f(y);
    if (Yb) Yb[(size_t)row * D + t] = f2bf(y);
    if (Yf) Yf[(size_t)row * D + t] = y;
  }
}

// adjacency bitmask: edges + diagonal
__global__ void k_build_adj(const long long* __restrict__ ei,
                            unsigned int* __restrict__ adj, int E_, int N_) {
  int i = blockIdx.x * 256 + threadIdx.x;
  if (i < E_) {
    int s = (int)ei[i], t = (int)ei[E_ + i];
    atomicOr(&adj[(size_t)s * (N_ / 32) + (t >> 5)], 1u << (t & 31));
  } else if (i < E_ + N_) {
    int d = i - E_;
    atomicOr(&adj[(size_t)d * (N_ / 32) + (d >> 5)], 1u << (d & 31));
  }
}

// scatter edge bias into dense scores S[h][src][tgt]
__global__ void k_bias_scatter(float* __restrict__ S,
                               const float* __restrict__ eb,
                               const long long* __restrict__ ei, int E_, int N_) {
  int e = blockIdx.x * 256 + threadIdx.x;
  if (e >= E_) return;
  int s = (int)ei[e], t = (int)ei[E_ + e];
  size_t nn = (size_t)N_ * N_;
#pragma unroll
  for (int h = 0; h < GHEADS; ++h)
    S[h * nn + (size_t)s * N_ + t] += eb[(size_t)e * GHEADS + h];
}

// masked softmax over keys (row of S) -> bf16 attention probs
__global__ __launch_bounds__(256) void k_masked_softmax(
    const float* __restrict__ S, const unsigned int* __restrict__ adj,
    unsigned short* __restrict__ Aout, int n)
{
  __shared__ float red[256];
  const int row = blockIdx.x, head = blockIdx.y, t = threadIdx.x;
  const size_t base = ((size_t)head * n + row) * n;
  const unsigned int* arow = adj + (size_t)row * (n / 32);
  float mx = -3.0e38f;
  for (int m = t; m < n; m += 256)
    if (arow[m >> 5] & (1u << (m & 31))) mx = fmaxf(mx, S[base + m]);
  red[t] = mx; __syncthreads();
  for (int s = 128; s > 0; s >>= 1) { if (t < s) red[t] = fmaxf(red[t], red[t + s]); __syncthreads(); }
  mx = red[0]; __syncthreads();
  float sum = 0.0f;
  for (int m = t; m < n; m += 256)
    if (arow[m >> 5] & (1u << (m & 31))) sum += __expf(S[base + m] - mx);
  red[t] = sum; __syncthreads();
  for (int s = 128; s > 0; s >>= 1) { if (t < s) red[t] += red[t + s]; __syncthreads(); }
  const float inv = 1.0f / red[0];
  for (int m = t; m < n; m += 256) {
    float v = (arow[m >> 5] & (1u << (m & 31))) ? __expf(S[base + m] - mx) * inv : 0.0f;
    Aout[base + m] = f2bf(v);
  }
}

// V fp32 [N,256] -> Vt bf16 [(h*32+d), N]  (B^T for O = A*V per head)
__global__ void k_vt(const float* __restrict__ V,
                     unsigned short* __restrict__ Vt, int N_) {
  int i = blockIdx.x * 256 + threadIdx.x;
  if (i >= N_ * GHID) return;
  int m = i >> 8, c = i & 255;
  Vt[(size_t)c * N_ + m] = f2bf(V[i]);
}

// acc = (init?0:acc) + w5[li]*x
__global__ void k_axpy(float* __restrict__ acc, const float* __restrict__ x,
                       const float* __restrict__ w5, int li, int n, int init) {
  int i = blockIdx.x * 256 + threadIdx.x;
  if (i < n) { float a = init ? 0.0f : acc[i]; acc[i] = a + w5[li] * x[i]; }
}

__global__ void k_softmax5(const float* __restrict__ lw, float* __restrict__ w5, int n) {
  if (threadIdx.x == 0 && blockIdx.x == 0) {
    float mx = -3.0e38f;
    for (int i = 0; i < n; ++i) mx = fmaxf(mx, lw[i]);
    float s = 0.0f;
    for (int i = 0; i < n; ++i) { float e = __expf(lw[i] - mx); w5[i] = e; s += e; }
    for (int i = 0; i < n; ++i) w5[i] /= s;
  }
}

// edge_repr[e, 0:576] = concat(x[src], x[tgt], edge_attr[e]) -> bf16
__global__ void k_erep(const float* __restrict__ xf, const float* __restrict__ eattr,
                       const long long* __restrict__ ei,
                       unsigned short* __restrict__ erep, int E_, int N_) {
  long long idx = (long long)blockIdx.x * 256 + threadIdx.x;
  if (idx >= (long long)E_ * 576) return;
  int e = (int)(idx / 576), c = (int)(idx % 576);
  float v;
  if (c < 256)      v = xf[(size_t)ei[e] * 256 + c];
  else if (c < 512) v = xf[(size_t)ei[E_ + e] * 256 + (c - 256)];
  else              v = eattr[(size_t)e * GEDGE + (c - 512)];
  erep[idx] = f2bf(v);
}

// ---------------------------------------------------------------------------
extern "C" void kernel_launch(void* const* d_in, const int* in_sizes, int n_in,
                              void* d_out, int out_size, void* d_ws, size_t ws_size,
                              hipStream_t stream)
{
  (void)in_sizes; (void)n_in; (void)out_size; (void)ws_size;
  // Input flattening assumption: top-level insertion order
  // (node_features, edge_attr, params-leaves, edge_index); params leaves in
  // jax tree order (alphabetical keys, depth-first, 'b' before 'g'/'w').
  const float*     nf    = (const float*)d_in[0];
  const float*     eattr = (const float*)d_in[1];
  const long long* ei    = (const long long*)d_in[95];     // [2, E] int64
  auto P = [&](int i) { return (const float*)d_in[i]; };
  // params leaf indices:
  //  2 cls.l1.b   3 cls.l1.w   4 cls.l2.b   5 cls.l2.w   6 cls.l3.b  7 cls.l3.w
  //  8 cls.n1.b   9 cls.n1.g  10 cls.n2.b  11 cls.n2.g
  // 12 exist.l1.b 13 exist.l1.w 14 exist.l2.b 15 exist.l2.w 16 exist.n1.b 17 exist.n1.g
  // 18 layer_weights ; layer l base L=19+18l:
  //  +0 edge.b +1 edge.w +2 ffn1.b +3 ffn1.w +4 ffn2.b +5 ffn2.w +6 k.b +7 k.w
  //  +8 n1.b +9 n1.g +10 n2.b +11 n2.g +12 out.b +13 out.w +14 q.b +15 q.w +16 v.b +17 v.w
  // 91 node_proj.lin.b 92 node_proj.lin.w 93 node_proj.ln.b 94 node_proj.ln.g

  // ---- workspace carve ----------------------------------------------------
  char* p = (char*)d_ws;
  auto carve = [&](size_t bytes) { void* r = (void*)p; p += (bytes + 255) & ~(size_t)255; return r; };
  float*          w5   = (float*)carve(5 * 4);
  unsigned int*   adj  = (unsigned int*)carve((size_t)GN * (GN / 32) * 4);
  unsigned short* wbuf = (unsigned short*)carve((size_t)262144 * 2);   // max W^T
  float*          xA   = (float*)carve((size_t)GN * GHID * 4);
  float*          xB   = (float*)carve((size_t)GN * GHID * 4);
  float*          acc  = (float*)carve((size_t)GN * GHID * 4);
  unsigned short* hb   = (unsigned short*)carve((size_t)GN * GHID * 2);
  unsigned short* qb   = (unsigned short*)carve((size_t)GN * GHID * 2);
  unsigned short* kb   = (unsigned short*)carve((size_t)GN * GHID * 2);
  float*          vf   = (float*)carve((size_t)GN * GHID * 4);
  unsigned short* vt   = (unsigned short*)carve((size_t)GN * GHID * 2);
  float*          ao   = (float*)carve((size_t)GN * GHID * 4);
  unsigned short* aob  = (unsigned short*)carve((size_t)GN * GHID * 2);
  float*          ebuf = (float*)carve((size_t)GE * GHEADS * 4);
  unsigned short* ffnb = (unsigned short*)carve((size_t)GN * 1024 * 2);
  unsigned short* eab  = (unsigned short*)carve((size_t)GE * GEDGE * 2);
  char*           big  = (char*)carve((size_t)192 << 20);  // scores+probs region
  float*          S    = (float*)big;                                   // 128 MB
  unsigned short* Abf  = (unsigned short*)(big + (size_t)GHEADS * GN * GN * 4); // 64 MB
  // edge-head aliases (run strictly after attention is done with `big`)
  unsigned short* erep = (unsigned short*)big;                          // 75.5 MB
  float*          h1f  = (float*)(big + ((size_t)80 << 20));            // 64 MB
  unsigned short* h1b  = (unsigned short*)(big + ((size_t)144 << 20));  // 32 MB
  float*          h2f  = (float*)h1f;                                   // 32 MB
  unsigned short* h2b  = (unsigned short*)(big + ((size_t)112 << 20));  // 16 MB
  float*          exf  = (float*)h1b;                                   // 32 MB
  unsigned short* exb  = (unsigned short*)h1f;                          // 16 MB

  auto gemm = [&](const unsigned short* A, int lda, unsigned long long sAz,
                  const unsigned short* Bt, int ldb, unsigned long long sBz,
                  float* Cf, unsigned short* Cb, int ldc, unsigned long long sCz,
                  const float* bias, const float* resid, float alpha,
                  int M, int Nc, int K, int flags, int nz) {
    dim3 grid((Nc + 31) / 32, (M + 255) / 256, nz);
    hipLaunchKernelGGL(k_gemm, grid, dim3(256), 0, stream, A, lda, sAz, Bt, ldb,
                       sBz, Cf, Cb, ldc, sCz, bias, resid, alpha, M, Nc, K, flags);
  };
  auto wconv = [&](const float* W, int K, int Nc) {
    int n = K * Nc;
    hipLaunchKernelGGL(k_wconv, dim3((n + 255) / 256), dim3(256), 0, stream, W, wbuf, K, Nc);
  };
  auto tobf = [&](const float* x, unsigned short* y, size_t n) {
    hipLaunchKernelGGL(k_f32bf16, dim3((unsigned)((n + 255) / 256)), dim3(256), 0, stream, x, y, n);
  };
  auto ln = [&](const float* X, const float* g, const float* b,
                unsigned short* Yb, float* Yf, int rows, int D, int dogelu) {
    hipLaunchKernelGGL(k_ln, dim3(rows), dim3(256), 0, stream, X, g, b, Yb, Yf, D, dogelu);
  };

  // ---- prologue -----------------------------------------------------------
  hipLaunchKernelGGL(k_softmax5, dim3(1), dim3(1), 0, stream, P(18), w5, GLAYERS + 1);
  hipMemsetAsync(adj, 0, (size_t)GN * (GN / 32) * 4, stream);
  hipLaunchKernelGGL(k_build_adj, dim3((GE + GN + 255) / 256), dim3(256), 0, stream, ei, adj, GE, GN);
  tobf(eattr, eab, (size_t)GE * GEDGE);

  // node projection: x0 = gelu(ln(nf @ W + b))
  tobf(nf, hb, (size_t)GN * GHID);
  wconv(P(92), GHID, GHID);
  gemm(hb, GHID, 0, wbuf, GHID, 0, xB, nullptr, GHID, 0, P(91), nullptr, 1.0f,
       GN, GHID, GHID, FBIAS | FF32, 1);
  ln(xB, P(94), P(93), nullptr, xA, GN, GHID, 1);
  hipLaunchKernelGGL(k_axpy, dim3((GN * GHID + 255) / 256), dim3(256), 0, stream,
                     acc, xA, w5, 0, GN * GHID, 1);

  const float inv_scale = 0.17677669529663687f;   // 1/sqrt(HEAD_DIM)

  // ---- transformer layers -------------------------------------------------
  for (int l = 0; l < GLAYERS; ++l) {
    const int L = 19 + 18 * l;
    ln(xA, P(L + 9), P(L + 8), hb, nullptr, GN, GHID, 0);           // norm1
    wconv(P(L + 15), GHID, GHID);                                   // Q
    gemm(hb, GHID, 0, wbuf, GHID, 0, nullptr, qb, GHID, 0, P(L + 14), nullptr,
         1.0f, GN, GHID, GHID, FBIAS | FBF16, 1);
    wconv(P(L + 7), GHID, GHID);                                    // K
    gemm(hb, GHID, 0, wbuf, GHID, 0, nullptr, kb, GHID, 0, P(L + 6), nullptr,
         1.0f, GN, GHID, GHID, FBIAS | FBF16, 1);
    wconv(P(L + 17), GHID, GHID);                                   // V
    gemm(hb, GHID, 0, wbuf, GHID, 0, vf, nullptr, GHID, 0, P(L + 16), nullptr,
         1.0f, GN, GHID, GHID, FBIAS | FF32, 1);
    hipLaunchKernelGGL(k_vt, dim3((GN * GHID + 255) / 256), dim3(256), 0, stream, vf, vt, GN);

    // S[h] = Q_h K_h^T / sqrt(d)   (one WMMA K-step: HEAD_DIM==32)
    gemm(qb, GHID, GHD, kb, GHID, GHD, S, nullptr, GN, (unsigned long long)GN * GN,
         nullptr, nullptr, inv_scale, GN, GN, GHD, FF32, GHEADS);

    wconv(P(L + 1), GEDGE, GHEADS);                                 // edge bias
    gemm(eab, GEDGE, 0, wbuf, GEDGE, 0, ebuf, nullptr, GHEADS, 0, P(L + 0),
         nullptr, 1.0f, GE, GHEADS, GEDGE, FBIAS | FF32, 1);
    hipLaunchKernelGGL(k_bias_scatter, dim3((GE + 255) / 256), dim3(256), 0,
                       stream, S, ebuf, ei, GE, GN);
    hipLaunchKernelGGL(k_masked_softmax, dim3(GN, GHEADS), dim3(256), 0, stream,
                       S, adj, Abf, GN);

    // O[h] = A_h V_h  (A bf16 straight from softmax)
    gemm(Abf, GN, (unsigned long long)GN * GN, vt, GN, (unsigned long long)GHD * GN,
         ao, nullptr, GHID, GHD, nullptr, nullptr, 1.0f, GN, GHD, GN, FF32, GHEADS);

    tobf(ao, aob, (size_t)GN * GHID);                               // out proj + res
    wconv(P(L + 13), GHID, GHID);
    gemm(aob, GHID, 0, wbuf, GHID, 0, xB, nullptr, GHID, 0, P(L + 12), xA, 1.0f,
         GN, GHID, GHID, FBIAS | FRES | FF32, 1);

    ln(xB, P(L + 11), P(L + 10), hb, nullptr, GN, GHID, 0);         // norm2 + FFN
    wconv(P(L + 3), GHID, 4 * GHID);
    gemm(hb, GHID, 0, wbuf, GHID, 0, nullptr, ffnb, 4 * GHID, 0, P(L + 2), nullptr,
         1.0f, GN, 4 * GHID, GHID, FBIAS | FGELU | FBF16, 1);
    wconv(P(L + 5), 4 * GHID, GHID);
    gemm(ffnb, 4 * GHID, 0, wbuf, 4 * GHID, 0, xA, nullptr, GHID, 0, P(L + 4), xB,
         1.0f, GN, GHID, 4 * GHID, FBIAS | FRES | FF32, 1);

    hipLaunchKernelGGL(k_axpy, dim3((GN * GHID + 255) / 256), dim3(256), 0,
                       stream, acc, xA, w5, l + 1, GN * GHID, 0);
  }

  // ---- edge heads ---------------------------------------------------------
  {
    long long tot = (long long)GE * 576;
    hipLaunchKernelGGL(k_erep, dim3((unsigned)((tot + 255) / 256)), dim3(256), 0,
                       stream, acc, eattr, ei, erep, GE, GN);
  }
  // cls: l1 -> ln+gelu -> l2 -> ln+gelu -> l3
  wconv(P(3), 576, 256);
  gemm(erep, 576, 0, wbuf, 576, 0, h1f, nullptr, 256, 0, P(2), nullptr, 1.0f,
       GE, 256, 576, FBIAS | FF32, 1);
  ln(h1f, P(9), P(8), h1b, nullptr, GE, 256, 1);
  wconv(P(5), 256, 128);
  gemm(h1b, 256, 0, wbuf, 256, 0, h2f, nullptr, 128, 0, P(4), nullptr, 1.0f,
       GE, 128, 256, FBIAS | FF32, 1);
  ln(h2f, P(11), P(10), h2b, nullptr, GE, 128, 1);
  wconv(P(7), 128, 8);
  gemm(h2b, 128, 0, wbuf, 128, 0, (float*)d_out, nullptr, 8, 0, P(6), nullptr,
       1.0f, GE, 8, 128, FBIAS | FF32, 1);
  // exist: l1 -> ln+gelu -> l2
  wconv(P(13), 576, 128);
  gemm(erep, 576, 0, wbuf, 576, 0, exf, nullptr, 128, 0, P(12), nullptr, 1.0f,
       GE, 128, 576, FBIAS | FF32, 1);
  ln(exf, P(17), P(16), exb, nullptr, GE, 128, 1);
  wconv(P(15), 128, 1);
  gemm(exb, 128, 0, wbuf, 128, 0, (float*)d_out + (size_t)GE * 8, nullptr, 1, 0,
       P(14), nullptr, 1.0f, GE, 1, 128, FBIAS | FF32, 1);
}
